// RecommendationModel_75247827026327
// MI455X (gfx1250) — compile-verified, matrix operand order checked by reading, not run on hardware
//
#include <hip/hip_runtime.h>
#include <hip/hip_bf16.h>

// ---------------------------------------------------------------------------
// GCN recommendation model for MI455X (gfx1250, wave32, WMMA).
// All dense math via v_wmma_f32_16x16x32_f16 (f16 in, f32 accumulate).
// ---------------------------------------------------------------------------

typedef __attribute__((ext_vector_type(16))) _Float16 v16h;
typedef __attribute__((ext_vector_type(8)))  float    v8f;

union V16HU { v16h v; unsigned int u[8]; };

#define N_NODES 100000
#define N_EDGES 1600000
#define N_QUERIES 200000
#define USER_OFFSET 30000   // NUM_PRODUCTS + NUM_INGREDIENTS
#define BN_EPS 1e-5f

// ---------------------------------------------------------------------------
// small utility kernels
// ---------------------------------------------------------------------------
__global__ void k_zero_f32(float* __restrict__ p, size_t n) {
    size_t i = (size_t)blockIdx.x * blockDim.x + threadIdx.x;
    if (i < n) p[i] = 0.0f;
}

__global__ void k_degree(const int* __restrict__ dst, float* __restrict__ deg, int E) {
    int i = blockIdx.x * blockDim.x + threadIdx.x;
    if (i < E) atomicAdd(&deg[dst[i]], 1.0f);
}

__global__ void k_dinv(float* __restrict__ d, int n) {
    int i = blockIdx.x * blockDim.x + threadIdx.x;
    if (i < n) d[i] = rsqrtf(d[i] + 1.0f);   // self-loop adds 1 to degree
}

__global__ void k_f32_to_f16(const float* __restrict__ x, _Float16* __restrict__ o, size_t n) {
    size_t i = (size_t)blockIdx.x * blockDim.x + threadIdx.x;
    if (i < n) o[i] = (_Float16)x[i];
}

// ---------------------------------------------------------------------------
// Node-feature GEMM:  out[N, Dout] (f32) = act[N, Din] (f16) @ W[Din, Dout] (f32->f16)
// Block = 128 threads = 4 waves; block covers 64 rows; wave w owns rows
// [64*bx + 16w, +16) and iterates over all Dout/16 column tiles, K in steps of 32.
// ---------------------------------------------------------------------------
__global__ void k_gemm_nodes(const _Float16* __restrict__ act,
                             const float* __restrict__ W,
                             float* __restrict__ out,
                             int nRows, int Din, int Dout) {
    __shared__ _Float16 sA[64 * 128];     // A stripe (row stride = Din)
    __shared__ _Float16 sW[128 * 128];    // W transposed: sW[n*Din + k] = W[k][n]

    const int tid  = threadIdx.x;
    const int row0 = blockIdx.x * 64;

    // stage A (already f16 in global) as dwords
    const int rowWords = Din >> 1;
    unsigned int* uA = (unsigned int*)sA;
    for (int i = tid; i < 64 * rowWords; i += 128) {
        int r = i / rowWords, c = i - r * rowWords;
        int gr = row0 + r;
        unsigned int v = 0u;
        if (gr < nRows)
            v = ((const unsigned int*)(act + (size_t)gr * Din))[c];
        uA[i] = v;
    }
    // stage W transposed, f32 -> f16
    for (int i = tid; i < Dout * Din; i += 128) {
        int n = i / Din, k = i - n * Din;
        sW[i] = (_Float16)W[(size_t)k * Dout + n];
    }
    __syncthreads();

    const int lane = tid & 31, wv = tid >> 5;
    const int hi   = lane >> 4;               // lane >= 16
    const int mrow = wv * 16 + (lane & 15);   // A row this lane feeds
    const int nTiles = Dout >> 4;

    for (int nt = 0; nt < nTiles; ++nt) {
        v8f acc = {};
        const int ncol = nt * 16 + (lane & 15);
        for (int kc = 0; kc < Din; kc += 32) {
            V16HU a, b;
#pragma unroll
            for (int j = 0; j < 8; ++j) {
                int kp = kc + ((j < 4) ? (2 * j) : (16 + 2 * (j - 4))) + (hi ? 8 : 0);
                a.u[j] = *(const unsigned int*)(&sA[mrow * Din + kp]);
                b.u[j] = *(const unsigned int*)(&sW[ncol * Din + kp]);
            }
            acc = __builtin_amdgcn_wmma_f32_16x16x32_f16(
                false, a.v, false, b.v, (short)0, acc, false, false);
        }
#pragma unroll
        for (int r = 0; r < 8; ++r) {
            int gr = row0 + wv * 16 + r + hi * 8;
            int gc = nt * 16 + (lane & 15);
            if (gr < nRows) out[(size_t)gr * Dout + gc] = acc[r];
        }
    }
}

// ---------------------------------------------------------------------------
// Edge scatter: agg[dst] += h[src] * dinv[src] * dinv[dst]   (per feature)
// thread -> (edge, feature); consecutive threads share an edge -> coalesced.
// ---------------------------------------------------------------------------
__global__ void k_edge_agg(const float* __restrict__ h,
                           const int* __restrict__ src,
                           const int* __restrict__ dst,
                           const float* __restrict__ dinv,
                           float* __restrict__ agg,
                           int E, int shift) {
    size_t idx = (size_t)blockIdx.x * blockDim.x + threadIdx.x;
    size_t total = (size_t)E << shift;
    if (idx >= total) return;
    int e = (int)(idx >> shift);
    int f = (int)(idx & (((size_t)1 << shift) - 1));
    int s = src[e], d = dst[e];
    float c = dinv[s] * dinv[d];
    atomicAdd(&agg[((size_t)d << shift) + f], h[((size_t)s << shift) + f] * c);
}

// ---------------------------------------------------------------------------
// GCN epilogue: v = agg + h*dinv^2 + bias, optional folded-BN, optional ReLU,
// emit f16 activations for the next GEMM.
// ---------------------------------------------------------------------------
__global__ void k_gcn_post(const float* __restrict__ agg,
                           const float* __restrict__ h,
                           const float* __restrict__ dinv,
                           const float* __restrict__ bias,
                           const float* __restrict__ gamma,
                           const float* __restrict__ beta,
                           const float* __restrict__ mean,
                           const float* __restrict__ var,
                           int useBn, int useRelu,
                           _Float16* __restrict__ outA,
                           int nRows, int shift) {
    size_t idx = (size_t)blockIdx.x * blockDim.x + threadIdx.x;
    size_t total = (size_t)nRows << shift;
    if (idx >= total) return;
    int i = (int)(idx >> shift);
    int f = (int)(idx & (((size_t)1 << shift) - 1));
    float dv = dinv[i];
    float v = agg[idx] + h[idx] * dv * dv + bias[f];
    if (useBn) {
        float s = gamma[f] * rsqrtf(var[f] + BN_EPS);
        v = (v - mean[f]) * s + beta[f];
    }
    if (useRelu) v = fmaxf(v, 0.0f);
    outA[idx] = (_Float16)v;
}

// ---------------------------------------------------------------------------
// Fused MLP head: gather [user|prod] embeddings -> 128 -> BN/ReLU -> 128
//                 -> 64 -> BN/ReLU -> 1 -> sigmoid*4+1.
// One block = 128 threads = 4 waves handles 64 queries; all weights in LDS.
// ---------------------------------------------------------------------------
__global__ void k_mlp_fused(const _Float16* __restrict__ emb,   // [N_NODES, 64] f16
                            const int* __restrict__ uidx,
                            const int* __restrict__ pidx,
                            const float* __restrict__ W1, const float* __restrict__ b1,
                            const float* __restrict__ g1, const float* __restrict__ be1,
                            const float* __restrict__ mu1, const float* __restrict__ va1,
                            const float* __restrict__ W2, const float* __restrict__ b2,
                            const float* __restrict__ g2, const float* __restrict__ be2,
                            const float* __restrict__ mu2, const float* __restrict__ va2,
                            const float* __restrict__ W3, const float* __restrict__ b3,
                            float* __restrict__ out, int nQ) {
    __shared__ _Float16 sA[64 * 128];     // gathered features, later H1
    __shared__ _Float16 sW[128 * 128];    // Wt1; later Wt2 (first half) + H2
    __shared__ float    sw3[64];

    const int tid = threadIdx.x;
    const int q0  = blockIdx.x * 64;

    // gather query features: cols 0..63 = user emb, 64..127 = product emb
    unsigned int* uA = (unsigned int*)sA;
    const unsigned int* uE = (const unsigned int*)emb;   // 32 dwords per node row
    for (int i = tid; i < 64 * 64; i += 128) {
        int q = i >> 6, c = i & 63;
        int row = (c < 32) ? (uidx[q0 + q] + USER_OFFSET) : pidx[q0 + q];
        uA[i] = uE[(size_t)row * 32 + (c & 31)];
    }
    // Wt1[n*128 + k] = W1[k][n]
    for (int i = tid; i < 128 * 128; i += 128) {
        int n = i >> 7, k = i & 127;
        sW[i] = (_Float16)W1[(size_t)k * 128 + n];
    }
    if (tid < 64) sw3[tid] = W3[tid];
    __syncthreads();

    const int lane = tid & 31, wv = tid >> 5;
    const int hi   = lane >> 4;
    const int mrow = wv * 16 + (lane & 15);

    // ---- layer 1: [64,128] @ [128,128] ----
    v8f acc1[8];
    for (int nt = 0; nt < 8; ++nt) {
        v8f acc = {};
        const int ncol = nt * 16 + (lane & 15);
        for (int kc = 0; kc < 128; kc += 32) {
            V16HU a, b;
#pragma unroll
            for (int j = 0; j < 8; ++j) {
                int kp = kc + ((j < 4) ? (2 * j) : (16 + 2 * (j - 4))) + (hi ? 8 : 0);
                a.u[j] = *(const unsigned int*)(&sA[mrow * 128 + kp]);
                b.u[j] = *(const unsigned int*)(&sW[ncol * 128 + kp]);
            }
            acc = __builtin_amdgcn_wmma_f32_16x16x32_f16(
                false, a.v, false, b.v, (short)0, acc, false, false);
        }
        acc1[nt] = acc;
    }
    __syncthreads();   // everyone done reading sA / sW

    // epilogue -> H1 (into sA); stage Wt2 into sW[0 .. 64*128)
    for (int nt = 0; nt < 8; ++nt) {
        int col = nt * 16 + (lane & 15);
        float s = g1[col] * rsqrtf(va1[col] + BN_EPS);
#pragma unroll
        for (int r = 0; r < 8; ++r) {
            float v = acc1[nt][r] + b1[col];
            v = (v - mu1[col]) * s + be1[col];
            v = fmaxf(v, 0.0f);
            int row = wv * 16 + r + hi * 8;
            sA[row * 128 + col] = (_Float16)v;
        }
    }
    for (int i = tid; i < 64 * 128; i += 128) {
        int n = i >> 7, k = i & 127;
        sW[i] = (_Float16)W2[(size_t)k * 64 + n];   // Wt2[n*128 + k]
    }
    __syncthreads();

    // ---- layer 2: [64,128] @ [128,64] ----  H2 stored after Wt2 region
    _Float16* sH2 = sW + 64 * 128;
    for (int nt = 0; nt < 4; ++nt) {
        v8f acc = {};
        const int ncol = nt * 16 + (lane & 15);
        for (int kc = 0; kc < 128; kc += 32) {
            V16HU a, b;
#pragma unroll
            for (int j = 0; j < 8; ++j) {
                int kp = kc + ((j < 4) ? (2 * j) : (16 + 2 * (j - 4))) + (hi ? 8 : 0);
                a.u[j] = *(const unsigned int*)(&sA[mrow * 128 + kp]);
                b.u[j] = *(const unsigned int*)(&sW[ncol * 128 + kp]);
            }
            acc = __builtin_amdgcn_wmma_f32_16x16x32_f16(
                false, a.v, false, b.v, (short)0, acc, false, false);
        }
        int col = nt * 16 + (lane & 15);
        float s = g2[col] * rsqrtf(va2[col] + BN_EPS);
#pragma unroll
        for (int r = 0; r < 8; ++r) {
            float v = acc[r] + b2[col];
            v = (v - mu2[col]) * s + be2[col];
            v = fmaxf(v, 0.0f);
            int row = wv * 16 + r + hi * 8;
            sH2[row * 64 + col] = (_Float16)v;
        }
    }
    __syncthreads();

    // ---- layer 3: 64-dim dot + sigmoid*4 + 1 ----
    if (tid < 64) {
        float s = 0.0f;
#pragma unroll
        for (int j = 0; j < 64; ++j) s += (float)sH2[tid * 64 + j] * sw3[j];
        s += b3[0];
        float r = 4.0f / (1.0f + __expf(-s)) + 1.0f;
        int q = q0 + tid;
        if (q < nQ) out[q] = r;
    }
}

// ---------------------------------------------------------------------------
// launch
// ---------------------------------------------------------------------------
static inline size_t alignUp(size_t x) { return (x + 255) & ~(size_t)255; }

extern "C" void kernel_launch(void* const* d_in, const int* in_sizes, int n_in,
                              void* d_out, int out_size, void* d_ws, size_t ws_size,
                              hipStream_t stream) {
    (void)in_sizes; (void)n_in; (void)out_size; (void)ws_size;

    const int N = N_NODES, E = N_EDGES, Q = N_QUERIES;

    // inputs (setup_inputs dict order, params flattened in insertion order)
    const float* x    = (const float*)d_in[0];
    const int*   ei   = (const int*)d_in[1];
    const int*   src  = ei;
    const int*   dst  = ei + E;
    const int*   uidx = (const int*)d_in[2];
    const int*   pidx = (const int*)d_in[3];
    const float* g1W  = (const float*)d_in[4];  const float* g1b = (const float*)d_in[5];
    const float* g2W  = (const float*)d_in[6];  const float* g2b = (const float*)d_in[7];
    const float* g3W  = (const float*)d_in[8];  const float* g3b = (const float*)d_in[9];
    const float* bn1g = (const float*)d_in[10]; const float* bn1b = (const float*)d_in[11];
    const float* bn1m = (const float*)d_in[12]; const float* bn1v = (const float*)d_in[13];
    const float* bn2g = (const float*)d_in[14]; const float* bn2b = (const float*)d_in[15];
    const float* bn2m = (const float*)d_in[16]; const float* bn2v = (const float*)d_in[17];
    const float* m1W  = (const float*)d_in[18]; const float* m1b = (const float*)d_in[19];
    const float* mb1g = (const float*)d_in[20]; const float* mb1b = (const float*)d_in[21];
    const float* mb1m = (const float*)d_in[22]; const float* mb1v = (const float*)d_in[23];
    const float* m2W  = (const float*)d_in[24]; const float* m2b = (const float*)d_in[25];
    const float* mb2g = (const float*)d_in[26]; const float* mb2b = (const float*)d_in[27];
    const float* mb2m = (const float*)d_in[28]; const float* mb2v = (const float*)d_in[29];
    const float* m3W  = (const float*)d_in[30]; const float* m3b = (const float*)d_in[31];
    float* out = (float*)d_out;

    // workspace layout
    char* ws = (char*)d_ws;
    size_t off = 0;
    float*    dinv = (float*)(ws + off);    off = alignUp(off + (size_t)N * 4);
    _Float16* act  = (_Float16*)(ws + off); off = alignUp(off + (size_t)N * 128 * 2);
    float*    hbuf = (float*)(ws + off);    off = alignUp(off + (size_t)N * 128 * 4);
    float*    agg  = (float*)(ws + off);    off = alignUp(off + (size_t)N * 128 * 4);
    _Float16* emb  = act;   // reuse: activations dead once layer-3 GEMM consumed them

    const int T = 256;
    auto blocks = [](size_t n, int t) { return (unsigned)((n + t - 1) / t); };

    // degree -> dinv
    k_zero_f32<<<blocks((size_t)N, T), T, 0, stream>>>(dinv, (size_t)N);
    k_degree<<<blocks((size_t)E, T), T, 0, stream>>>(dst, dinv, E);
    k_dinv<<<blocks((size_t)N, T), T, 0, stream>>>(dinv, N);

    // x -> f16
    k_f32_to_f16<<<blocks((size_t)N * 64, T), T, 0, stream>>>(x, act, (size_t)N * 64);

    const unsigned gemmBlocks = (N + 63) / 64;

    // ---- GCN layer 1: 64 -> 128, BN1 + ReLU ----
    k_gemm_nodes<<<gemmBlocks, 128, 0, stream>>>(act, g1W, hbuf, N, 64, 128);
    k_zero_f32<<<blocks((size_t)N * 128, T), T, 0, stream>>>(agg, (size_t)N * 128);
    k_edge_agg<<<blocks((size_t)E << 7, T), T, 0, stream>>>(hbuf, src, dst, dinv, agg, E, 7);
    k_gcn_post<<<blocks((size_t)N << 7, T), T, 0, stream>>>(
        agg, hbuf, dinv, g1b, bn1g, bn1b, bn1m, bn1v, 1, 1, act, N, 7);

    // ---- GCN layer 2: 128 -> 128, BN2 + ReLU ----
    k_gemm_nodes<<<gemmBlocks, 128, 0, stream>>>(act, g2W, hbuf, N, 128, 128);
    k_zero_f32<<<blocks((size_t)N * 128, T), T, 0, stream>>>(agg, (size_t)N * 128);
    k_edge_agg<<<blocks((size_t)E << 7, T), T, 0, stream>>>(hbuf, src, dst, dinv, agg, E, 7);
    k_gcn_post<<<blocks((size_t)N << 7, T), T, 0, stream>>>(
        agg, hbuf, dinv, g2b, bn2g, bn2b, bn2m, bn2v, 1, 1, act, N, 7);

    // ---- GCN layer 3: 128 -> 64 (no BN / ReLU) -> embeddings (f16) ----
    k_gemm_nodes<<<gemmBlocks, 128, 0, stream>>>(act, g3W, hbuf, N, 128, 64);
    k_zero_f32<<<blocks((size_t)N * 64, T), T, 0, stream>>>(agg, (size_t)N * 64);
    k_edge_agg<<<blocks((size_t)E << 6, T), T, 0, stream>>>(hbuf, src, dst, dinv, agg, E, 6);
    k_gcn_post<<<blocks((size_t)N << 6, T), T, 0, stream>>>(
        agg, hbuf, dinv, g3b, nullptr, nullptr, nullptr, nullptr, 0, 0, emb, N, 6);

    // ---- fused MLP head ----
    k_mlp_fused<<<(Q + 63) / 64, 128, 0, stream>>>(
        emb, uidx, pidx,
        m1W, m1b, mb1g, mb1b, mb1m, mb1v,
        m2W, m2b, mb2g, mb2b, mb2m, mb2v,
        m3W, m3b, out, Q);
}
